// Polynomial_Network_15522011808086
// MI455X (gfx1250) — matrix operand order repaired and checked by reading, not compiled, not run on hardware
//
#include <hip/hip_runtime.h>

// ---------------------------------------------------------------------------
// Polynomial network:  h_{l+1}[b,n] = [h_l,1]^T W_n [h_l,1]  (3 layers) + linear
// Strategy: one big GEMM per layer, M=2048(batch) x N=256(neurons) x K=257*257,
// with A[b,(i,j)] = z[b,i]*z[b,j] generated on the fly from an LDS-staged Z tile
// and B[(i,j),n] = W[n][i][j] pre-split into bf16 hi/lo (3-product split-bf16
// emulation => ~fp32-grade accuracy on the WMMA pipe).
// ---------------------------------------------------------------------------

typedef __bf16  v16bf __attribute__((ext_vector_type(16)));
typedef float   v8f   __attribute__((ext_vector_type(8)));

union BF16x16 { v16bf v; unsigned short u[16]; };

// Split fp32 into truncated-bf16 hi and truncated-bf16(lo remainder).
// Pure integer ops (no cvt needed); combined representation error ~2^-16 rel.
__device__ __forceinline__ void split_bf(float a, unsigned short& hi, unsigned short& lo) {
    unsigned u = __builtin_bit_cast(unsigned, a);
    hi = (unsigned short)(u >> 16);
    float hf = __builtin_bit_cast(float, u & 0xffff0000u);
    float r  = a - hf;
    lo = (unsigned short)(__builtin_bit_cast(unsigned, r) >> 16);
}

// ---------------- geometry / workspace layout ----------------
#define BATCH   2048
#define NN      256      // neurons per layer
#define DP1     257      // d+1
#define ZLD     292      // padded Z row stride (floats): 288 logical + bank pad
#define KSPLIT  4
#define PSZ     ((size_t)BATCH * NN)          // one partial buffer, elements

#define OFF_Z1   ((size_t)0)
#define SZ_Z     ((size_t)BATCH * ZLD * 4)                 // 2,392,064
#define OFF_Z2   (OFF_Z1 + SZ_Z)
#define OFF_Z3   (OFF_Z2 + SZ_Z)
#define OFF_H4   (OFF_Z3 + SZ_Z)
#define OFF_P    (OFF_H4 + (size_t)BATCH * NN * 4)
#define OFF_WHI  (OFF_P  + (size_t)KSPLIT * BATCH * NN * 4)
#define SZ_WHL   ((size_t)NN * DP1 * 256 * 2)              // 33,685,504
#define OFF_WLO  (OFF_WHI + SZ_WHL)
#define OFF_WCT  (OFF_WLO + SZ_WHL)                        // 288*256 floats

// ---------------- prep: build Z1 from x, init pad columns of Z2/Z3, zero Wct tail
__global__ void prep_pn(const float* __restrict__ x, float* __restrict__ Z1,
                        float* __restrict__ Z2, float* __restrict__ Z3,
                        float* __restrict__ Wct) {
    const int b = blockIdx.x;      // 2048
    const int t = threadIdx.x;     // 128
    for (int j = t; j < ZLD; j += 128) {
        float v = (j < 256) ? x[(size_t)b * 256 + j] : ((j == 256) ? 1.0f : 0.0f);
        Z1[(size_t)b * ZLD + j] = v;
        if (j >= 256) {
            float w = (j == 256) ? 1.0f : 0.0f;
            Z2[(size_t)b * ZLD + j] = w;
            Z3[(size_t)b * ZLD + j] = w;
        }
    }
    if (b < 31) {  // zero Wct rows 257..287 (padded i range)
        for (int n = t; n < 256; n += 128)
            Wct[(size_t)(DP1 + b) * 256 + n] = 0.0f;
    }
}

// ---------------- W conversion: split W[n][i][0..255] -> bf16 hi/lo; Wct[i][n]=W[n][i][256]
__global__ void convw_pn(const float* __restrict__ W, __bf16* __restrict__ Whi,
                         __bf16* __restrict__ Wlo, float* __restrict__ Wct) {
    const int ni = blockIdx.x;                  // n*257 + i  (65792 blocks)
    const float* src = W + (size_t)ni * DP1;
    __bf16* dh = Whi + (size_t)ni * 256;
    __bf16* dl = Wlo + (size_t)ni * 256;
    const int j0 = threadIdx.x * 4;             // 64 threads * 4
#pragma unroll
    for (int e = 0; e < 4; ++e) {
        unsigned short hi, lo;
        split_bf(src[j0 + e], hi, lo);
        dh[j0 + e] = __builtin_bit_cast(__bf16, hi);
        dl[j0 + e] = __builtin_bit_cast(__bf16, lo);
    }
    if (threadIdx.x == 0) {
        const int n = ni / DP1, i = ni - n * DP1;
        Wct[(size_t)i * 256 + n] = src[256];
    }
}

// ---------------- main GEMM: one wave per block, 16(M) x 64(N) tile, split-K=4
__global__ void __launch_bounds__(32)
gemm_pn(const float* __restrict__ Z, const __bf16* __restrict__ Whi,
        const __bf16* __restrict__ Wlo, const float* __restrict__ Wct,
        float* __restrict__ P) {
    const int lane = threadIdx.x;
    const int r = lane & 15;       // A: row, B: n-column, C: n-column
    const int h = lane >> 4;       // K-half selector
    const int m0 = blockIdx.x * 16;
    const int n0 = blockIdx.y * 64;
    const int s  = blockIdx.z;

    __shared__ __align__(16) float zs[16 * ZLD];
    // stage 16 Z rows (292 floats each) into LDS, vectorized
    for (int row = 0; row < 16; ++row) {
        const float4* src = (const float4*)(Z + (size_t)(m0 + row) * ZLD);
        float4* dst = (float4*)(zs + row * ZLD);
        for (int c = lane; c < ZLD / 4; c += 32) dst[c] = src[c];
    }
    __syncthreads();

    v8f C[4] = {};   // 4 N-fragments of 16x16 f32

    const int iBeg = s * 65;
    const int iEnd = (iBeg + 65 < DP1) ? (iBeg + 65) : DP1;

    for (int i = iBeg; i < iEnd; ++i) {
        const float zi = zs[r * ZLD + i];
        size_t wbase[4];
#pragma unroll
        for (int nf = 0; nf < 4; ++nf)
            wbase[nf] = ((size_t)(n0 + nf * 16 + r) * DP1 + i) * 256;

        for (int jt = 0; jt < 8; ++jt) {
            const int j0 = jt * 32;
            // A fragment: A[r, k] = zi * z[r, j0+k], split hi/lo.
            // 16-bit A 16x32 layout (contiguous, matches B): K=h*16+e
            BF16x16 Ah, Al;
            const float* zp0 = zs + r * ZLD + j0 + h * 16;
            const float* zp1 = zp0 + 8;
#pragma unroll
            for (int e = 0; e < 8; ++e) {
                split_bf(zi * zp0[e], Ah.u[e],     Al.u[e]);
                split_bf(zi * zp1[e], Ah.u[8 + e], Al.u[8 + e]);
            }
#pragma unroll
            for (int nf = 0; nf < 4; ++nf) {
                const size_t off = wbase[nf] + j0 + h * 16;   // B: lane=n, K=h*16+e
                v16bf Bh = *(const v16bf*)(Whi + off);
                v16bf Bl = *(const v16bf*)(Wlo + off);
                C[nf] = __builtin_amdgcn_wmma_f32_16x16x32_bf16(false, Ah.v, false, Bh, (short)0, C[nf], false, false);
                C[nf] = __builtin_amdgcn_wmma_f32_16x16x32_bf16(false, Ah.v, false, Bl, (short)0, C[nf], false, false);
                C[nf] = __builtin_amdgcn_wmma_f32_16x16x32_bf16(false, Al.v, false, Bh, (short)0, C[nf], false, false);
            }
        }
    }

    if (s == 0) {
        // j==256 column: out += sum_i z[b,i] * W[n][i][256]; K over i, padded to 288
        for (int it = 0; it < 9; ++it) {
            const int i0 = it * 32;
            BF16x16 Ah, Al;
            const float* zp0 = zs + r * ZLD + i0 + h * 16;
            const float* zp1 = zp0 + 8;
#pragma unroll
            for (int e = 0; e < 8; ++e) {
                split_bf(zp0[e], Ah.u[e],     Al.u[e]);
                split_bf(zp1[e], Ah.u[8 + e], Al.u[8 + e]);
            }
#pragma unroll
            for (int nf = 0; nf < 4; ++nf) {
                const int ng = n0 + nf * 16 + r;
                BF16x16 Bh, Bl;
#pragma unroll
                for (int e = 0; e < 16; ++e) {
                    const int k = h * 16 + e;
                    split_bf(Wct[(size_t)(i0 + k) * 256 + ng], Bh.u[e], Bl.u[e]);
                }
                C[nf] = __builtin_amdgcn_wmma_f32_16x16x32_bf16(false, Ah.v, false, Bh.v, (short)0, C[nf], false, false);
                C[nf] = __builtin_amdgcn_wmma_f32_16x16x32_bf16(false, Ah.v, false, Bl.v, (short)0, C[nf], false, false);
                C[nf] = __builtin_amdgcn_wmma_f32_16x16x32_bf16(false, Al.v, false, Bh.v, (short)0, C[nf], false, false);
            }
        }
    }

    // store partials: C layout -> row = m0 + h*8 + v, col = n0 + nf*16 + r
    float* Pout = P + (size_t)s * PSZ;
#pragma unroll
    for (int nf = 0; nf < 4; ++nf) {
        const int ng = n0 + nf * 16 + r;
#pragma unroll
        for (int v = 0; v < 8; ++v) {
            const int b = m0 + h * 8 + v;
            Pout[(size_t)b * NN + ng] = C[nf][v];
        }
    }
}

// ---------------- deterministic split-K reduce; writes into next layer's padded Z (or H4)
__global__ void reduce_pn(const float* __restrict__ P, float* __restrict__ dst, int stride) {
    const int b = blockIdx.x;     // 2048
    const int n = threadIdx.x;    // 256
    const size_t idx = (size_t)b * NN + n;
    float v = P[idx] + P[idx + PSZ] + P[idx + 2 * PSZ] + P[idx + 3 * PSZ];
    dst[(size_t)b * stride + n] = v;
}

// ---------------- final linear: out[b] = h3[b,:] . w_out + b_out
__global__ void final_pn(const float* __restrict__ H4, const float* __restrict__ wout,
                         const float* __restrict__ bout, float* __restrict__ out) {
    const int b = blockIdx.x * blockDim.x + threadIdx.x;
    if (b >= BATCH) return;
    float s = bout[0];
    const float* hp = H4 + (size_t)b * NN;
    for (int n = 0; n < NN; ++n) s += hp[n] * wout[n];
    out[b] = s;
}

extern "C" void kernel_launch(void* const* d_in, const int* in_sizes, int n_in,
                              void* d_out, int out_size, void* d_ws, size_t ws_size,
                              hipStream_t stream) {
    const float* x    = (const float*)d_in[0];
    const float* W1   = (const float*)d_in[1];
    const float* W2   = (const float*)d_in[2];
    const float* W3   = (const float*)d_in[3];
    const float* wout = (const float*)d_in[4];
    const float* bout = (const float*)d_in[5];
    float* out = (float*)d_out;

    char* ws = (char*)d_ws;
    float*  Z1  = (float*)(ws + OFF_Z1);
    float*  Z2  = (float*)(ws + OFF_Z2);
    float*  Z3  = (float*)(ws + OFF_Z3);
    float*  H4  = (float*)(ws + OFF_H4);
    float*  P   = (float*)(ws + OFF_P);
    __bf16* Whi = (__bf16*)(ws + OFF_WHI);
    __bf16* Wlo = (__bf16*)(ws + OFF_WLO);
    float*  Wct = (float*)(ws + OFF_WCT);

    const dim3 ggrid(BATCH / 16, NN / 64, KSPLIT);   // 128 x 4 x 4 single-wave WGs

    prep_pn<<<BATCH, 128, 0, stream>>>(x, Z1, Z2, Z3, Wct);

    // layer 1
    convw_pn<<<NN * DP1, 64, 0, stream>>>(W1, Whi, Wlo, Wct);
    gemm_pn<<<ggrid, 32, 0, stream>>>(Z1, Whi, Wlo, Wct, P);
    reduce_pn<<<BATCH, NN, 0, stream>>>(P, Z2, ZLD);
    // layer 2
    convw_pn<<<NN * DP1, 64, 0, stream>>>(W2, Whi, Wlo, Wct);
    gemm_pn<<<ggrid, 32, 0, stream>>>(Z2, Whi, Wlo, Wct, P);
    reduce_pn<<<BATCH, NN, 0, stream>>>(P, Z3, ZLD);
    // layer 3
    convw_pn<<<NN * DP1, 64, 0, stream>>>(W3, Whi, Wlo, Wct);
    gemm_pn<<<ggrid, 32, 0, stream>>>(Z3, Whi, Wlo, Wct, P);
    reduce_pn<<<BATCH, NN, 0, stream>>>(P, H4, NN);

    final_pn<<<BATCH / 128, 128, 0, stream>>>(H4, wout, bout, out);
}
